// MLP_Bank_11098195493452
// MI455X (gfx1250) — matrix operand order repaired and checked by reading, not compiled, not run on hardware
//
#include <hip/hip_runtime.h>
#include <hip/hip_bf16.h>
#include <math.h>

// ---------- vector / WMMA types ----------
typedef __attribute__((ext_vector_type(2)))  float    f32x2;
typedef __attribute__((ext_vector_type(4)))  float    f32x4;
typedef __attribute__((ext_vector_type(4)))  unsigned u32x4;
typedef __attribute__((ext_vector_type(8)))  float    v8f;
typedef __attribute__((ext_vector_type(2)))  __bf16   v2bf;
typedef __attribute__((ext_vector_type(4)))  __bf16   v4bf;
typedef __attribute__((ext_vector_type(16))) __bf16   v16bf;

union AB {                 // WMMA 16-bit A/B operand (8 VGPRs)
    v16bf v;
    u32x4 q[2];
};
union C8 {                 // WMMA f32 C/D operand (8 VGPRs)
    v8f   v;
    f32x4 q[2];
};

// f32 pair -> packed bf16 (RNE). Vector fptrunc lets the backend use the
// native bf16 convert path on gfx1250 instead of a manual bit sequence.
__device__ __forceinline__ unsigned pack_bf16(float a, float b) {
    f32x2 t; t.x = a; t.y = b;
    v2bf p = __builtin_convertvector(t, v2bf);
    return __builtin_bit_cast(unsigned, p);
}

// Fast GELU (tanh form). v_tanh_f32 confirmed on this toolchain (TRANS pipe,
// co-executes with VALU/WMMA); exp2/rcp fallback kept for safety.
__device__ __forceinline__ float gelu_fast(float x) {
    const float u = x * x;
    // 0.7978845608*(1 + 0.044715 u) = fma(0.0356774081, u, 0.7978845608)
    const float t = x * __builtin_fmaf(0.035677408136300125f, u, 0.7978845608028654f);
#if __has_builtin(__builtin_amdgcn_tanhf)
    const float th = __builtin_amdgcn_tanhf(t);
#elif __has_builtin(__builtin_amdgcn_tanh_f32)
    const float th = __builtin_amdgcn_tanh_f32(t);
#else
    const float e  = __builtin_amdgcn_exp2f(t * -2.885390081777927f);
    const float th = __builtin_fmaf(2.0f, __builtin_amdgcn_rcpf(1.0f + e), -1.0f);
#endif
    const float h = 0.5f * x;
    return __builtin_fmaf(h, th, h);
}

__device__ __forceinline__ float gelu_exact(float x) {
    return 0.5f * x * (1.0f + erff(x * 0.70710678118654752f));
}

// ---------- constants ----------
#define CIN      64
#define NPB      65536          // pixels per batch (256*256)
#define PIXC     128            // pixels per chunk per block
#define XSTR     72             // bf16 stride of xs rows (144B, 16B aligned)
#define HSTR     264            // bf16 stride of hs rows (528B, 16B aligned)
#define SUMHF    448            // 64 + 128 + 256
#define NT1      56             // W1 tiles: 28 mt * 2 kt
#define NT2      56             // W2 tiles: 4*(2+4+8)

// LDS byte offsets
#define OFF_XS   0
#define OFF_HS   (OFF_XS + PIXC * XSTR * 2)      // 18432
#define OFF_W1   (OFF_HS + PIXC * HSTR * 2)      // 86016
#define OFF_W2   (OFF_W1 + NT1 * 512 * 2)        // 143360
#define OFF_B1   (OFF_W2 + NT2 * 512 * 2)        // 200704
#define OFF_B2   (OFF_B1 + SUMHF * 4)            // 202496
#define LDS_BYTES (OFF_B2 + 64 * 4)              // 202752 (198 KB < 320 KB WGP)

struct Params {
    const float* x;
    const float* w1[3];
    const float* b1[3];
    const float* w2[3];
    const float* b2[3];
    const float* logit;    // d_out tail, written by router_kernel
    float*       out;
};

// A-operand tile load: tile pre-swizzled so each lane reads its 16 contiguous
// bf16 (32B) -> 2x ds_load_b128.
__device__ __forceinline__ void loadA(const unsigned short* base, int tile, int lane, AB& a) {
    const unsigned short* ptr = base + tile * 512 + lane * 16;
    a.q[0] = *(const u32x4*)(ptr);
    a.q[1] = *(const u32x4*)(ptr + 8);
}

// =========================================================================
// Kernel A: per-(b,c) mean over H*W
// =========================================================================
__global__ void mean_kernel(const float* __restrict__ x, float* __restrict__ means) {
    __shared__ float red[256];
    const int bc = blockIdx.x;                       // b*64 + c
    const float* p = x + (size_t)bc * NPB;
    float s = 0.f;
#pragma unroll 4
    for (int j = 0; j < 64; ++j) {
        f32x4 v = *(const f32x4*)(p + (size_t)(j * 256 + threadIdx.x) * 4);
        s += v.x + v.y + v.z + v.w;
    }
    red[threadIdx.x] = s;
    __syncthreads();
    for (int off = 128; off > 0; off >>= 1) {
        if ((int)threadIdx.x < off) red[threadIdx.x] += red[threadIdx.x + off];
        __syncthreads();
    }
    if (threadIdx.x == 0) means[bc] = red[0] * (1.0f / 65536.0f);
}

// =========================================================================
// Kernel B: router (hidden_new -> out_tail[0:64], logit -> out_tail[64:76])
// =========================================================================
__global__ void router_kernel(const float* __restrict__ means,
                              const float* __restrict__ hidden,
                              const float* __restrict__ rw1, const float* __restrict__ rb1,
                              const float* __restrict__ rw3, const float* __restrict__ rb3,
                              float* __restrict__ out_tail) {
    __shared__ float hn[64];
    const int t = threadIdx.x;
    if (t < 64) {
        const int bb = t >> 4, j = t & 15;
        float s = rb1[j];
        const float* w = rw1 + j * 80;
        for (int k = 0; k < 64; ++k) s += w[k] * means[bb * 64 + k];
        for (int k = 0; k < 16; ++k) s += w[64 + k] * hidden[bb * 16 + k];
        const float g = gelu_exact(s);                // exact erf: tiny & visible output
        hn[t] = g;
        out_tail[t] = g;                              // hidden_new (b,16)
    }
    __syncthreads();
    if (t < 12) {
        const int bb = t / 3, e = t % 3;
        float s = rb3[e];
        for (int j = 0; j < 16; ++j) s += rw3[e * 16 + j] * hn[bb * 16 + j];
        out_tail[64 + t] = s > 0.f ? s : 0.f;          // relu logit (b,3)
    }
}

// =========================================================================
// Kernel C: fused 3-expert MLP via bf16 WMMA, f32 accumulation
// =========================================================================
__global__ __launch_bounds__(256) void expert_kernel(Params P) {
    extern __shared__ char smem[];
    unsigned short* xs   = (unsigned short*)(smem + OFF_XS);   // [128][72]
    unsigned short* hs   = (unsigned short*)(smem + OFF_HS);   // [128][264]
    unsigned short* w1sw = (unsigned short*)(smem + OFF_W1);   // 56 tiles * 512 bf16
    unsigned short* w2sw = (unsigned short*)(smem + OFF_W2);   // 56 tiles * 512 bf16
    float*          b1s  = (float*)(smem + OFF_B1);            // [448]
    float*          b2s  = (float*)(smem + OFF_B2);            // [64]

    const int tid = threadIdx.x;
    const int b   = blockIdx.x >> 7;          // 128 blocks per batch
    const int blk = blockIdx.x & 127;

    float lg[3];
    lg[0] = P.logit[b * 3 + 0];
    lg[1] = P.logit[b * 3 + 1];
    lg[2] = P.logit[b * 3 + 2];

    // ---- stage W1 pre-swizzled into WMMA A-layout tiles -----------------
    // tile t = mtG*2+kt; u32 slot r = L*8+j holds elements (i=2j, 2j+1):
//   row = mtG*16 + (L&15),  K = kt*32 + 16h + 2j, h=L>>4
    {
        unsigned* w1d = (unsigned*)w1sw;
        for (int idx = tid; idx < NT1 * 256; idx += 256) {
            const int t = idx >> 8, r = idx & 255;
            const int L = r >> 3, j = r & 7, h = L >> 4;
            const int mtG = t >> 1, kt = t & 1;
            const int row = mtG * 16 + (L & 15);
            const int K = kt * 32 + 16 * h + 2 * j;
            const int e    = (row < 64) ? 0 : ((row < 192) ? 1 : 2);
            const int base = (e == 0) ? 0 : ((e == 1) ? 64 : 192);
            const float* src = P.w1[e] + (row - base) * 64 + K;
            w1d[idx] = pack_bf16(src[0], src[1]);
        }
    }
    // ---- stage W2 (pre-scaled by logit) into A-layout tiles -------------
    // expert tile bases {0, 8, 24}; within expert: tile = mt*nkt + kt
    {
        unsigned* w2d = (unsigned*)w2sw;
        for (int idx = tid; idx < NT2 * 256; idx += 256) {
            const int t = idx >> 8, r = idx & 255;
            const int L = r >> 3, j = r & 7, h = L >> 4;
            int e, toff, hfe;
            if (t < 8)       { e = 0; toff = 0;  hfe = 64;  }
            else if (t < 24) { e = 1; toff = 8;  hfe = 128; }
            else             { e = 2; toff = 24; hfe = 256; }
            const int lt = t - toff;
            const int nkt = hfe >> 5;
            const int mt = lt / nkt, kt = lt % nkt;
            const int row = mt * 16 + (L & 15);              // out channel
            const int K = kt * 32 + 16 * h + 2 * j;
            const float* src = P.w2[e] + row * hfe + K;
            w2d[idx] = pack_bf16(src[0] * lg[e], src[1] * lg[e]);
        }
    }
    for (int idx = tid; idx < SUMHF; idx += 256) {
        const int e    = (idx < 64) ? 0 : ((idx < 192) ? 1 : 2);
        const int base = (e == 0) ? 0 : ((e == 1) ? 64 : 192);
        b1s[idx] = P.b1[e][idx - base];
    }
    if (tid < 64)
        b2s[tid] = lg[0] * P.b2[0][tid] + lg[1] * P.b2[1][tid] + lg[2] * P.b2[2][tid];
    __syncthreads();

    const int wave = tid >> 5, lane = tid & 31;
    const int col = lane & 15, half = lane >> 4;
    const int p = wave * 16 + col;                    // this lane's pixel row in LDS
    const size_t xbase = (size_t)b * CIN * NPB;

    for (int ch = blk; ch < 512; ch += 128) {
        const int pb = ch * PIXC;                     // pixel base within batch

        // ---- stage x tile -> xs[pixel][channel] (bf16) ----
        for (int idx = tid; idx < CIN * (PIXC / 4); idx += 256) {
            const int c  = idx >> 5;
            const int p4 = (idx & 31) << 2;
            f32x4 v = *(const f32x4*)(P.x + xbase + (size_t)c * NPB + pb + p4);
            v4bf bv = __builtin_convertvector(v, v4bf);
            xs[(p4 + 0) * XSTR + c] = __builtin_bit_cast(unsigned short, bv[0]);
            xs[(p4 + 1) * XSTR + c] = __builtin_bit_cast(unsigned short, bv[1]);
            xs[(p4 + 2) * XSTR + c] = __builtin_bit_cast(unsigned short, bv[2]);
            xs[(p4 + 3) * XSTR + c] = __builtin_bit_cast(unsigned short, bv[3]);
        }
        __syncthreads();

        // ---- hoisted GEMM1 B tiles (x, K=64 -> 2 tiles), shared by all experts
        AB bx0, bx1;
        {
            const unsigned short* xr = xs + p * XSTR + 16 * half;
            bx0.q[0] = *(const u32x4*)(xr);
            bx0.q[1] = *(const u32x4*)(xr + 8);
            bx1.q[0] = *(const u32x4*)(xr + 32);
            bx1.q[1] = *(const u32x4*)(xr + 40);
        }

        // ---- output accumulators, initialized with combined bias --------
        C8 oc[4];
#pragma unroll
        for (int mt = 0; mt < 4; ++mt) {
            const float* bp = b2s + mt * 16 + 8 * half;
            oc[mt].q[0] = *(const f32x4*)(bp);
            oc[mt].q[1] = *(const f32x4*)(bp + 4);
        }

#pragma unroll
        for (int e = 0; e < 3; ++e) {
            const int hfe   = (e == 0) ? 64 : ((e == 1) ? 128 : 256);
            const int rOff  = (e == 0) ? 0 : ((e == 1) ? 64 : 192);
            const int t1Off = rOff / 8;                       // W1 tile base
            const int t2Off = (e == 0) ? 0 : ((e == 1) ? 8 : 24);
            const int nkt   = hfe / 32;

            // ---- GEMM1: h1 = gelu(W1 x + b1) -> hs[pixel][k] (bf16) -----
#pragma unroll
            for (int mt = 0; mt < hfe / 16; ++mt) {
                C8 acc;
                const float* bp = b1s + rOff + mt * 16 + 8 * half;
                acc.q[0] = *(const f32x4*)(bp);               // C := bias
                acc.q[1] = *(const f32x4*)(bp + 4);
                AB a;
                loadA(w1sw, t1Off + mt * 2 + 0, lane, a);
                acc.v = __builtin_amdgcn_wmma_f32_16x16x32_bf16(
                    false, a.v, false, bx0.v, (short)0, acc.v, false, false);
                loadA(w1sw, t1Off + mt * 2 + 1, lane, a);
                acc.v = __builtin_amdgcn_wmma_f32_16x16x32_bf16(
                    false, a.v, false, bx1.v, (short)0, acc.v, false, false);
                const int kb = mt * 16 + 8 * half;            // D rows this lane holds
#pragma unroll
                for (int j = 0; j < 4; ++j) {
                    const float g0 = gelu_fast(acc.v[2 * j]);
                    const float g1 = gelu_fast(acc.v[2 * j + 1]);
                    *(unsigned*)(hs + p * HSTR + kb + 2 * j) = pack_bf16(g0, g1);
                }
            }

            // ---- GEMM2 (kt-outer, B tile reused over 4 row tiles) -------
#pragma unroll
            for (int kt = 0; kt < nkt; ++kt) {
                AB bh;
                const unsigned short* hr = hs + p * HSTR + kt * 32 + 16 * half;
                bh.q[0] = *(const u32x4*)(hr);
                bh.q[1] = *(const u32x4*)(hr + 8);
#pragma unroll
                for (int mt = 0; mt < 4; ++mt) {
                    AB a;
                    loadA(w2sw, t2Off + mt * nkt + kt, lane, a);
                    oc[mt].v = __builtin_amdgcn_wmma_f32_16x16x32_bf16(
                        false, a.v, false, bh.v, (short)0, oc[mt].v, false, false);
                }
            }
        }

        // ---- epilogue: bias already folded in, just store fp32 ----------
#pragma unroll
        for (int mt = 0; mt < 4; ++mt) {
#pragma unroll
            for (int v = 0; v < 8; ++v) {
                const int c = mt * 16 + 8 * half + v;
                P.out[((size_t)b * CIN + c) * NPB + pb + wave * 16 + col] = oc[mt].v[v];
            }
        }
        __syncthreads();   // xs/hs reused next chunk
    }
}

// =========================================================================
// host-side launcher
// =========================================================================
extern "C" void kernel_launch(void* const* d_in, const int* in_sizes, int n_in,
                              void* d_out, int out_size, void* d_ws, size_t ws_size,
                              hipStream_t stream) {
    (void)in_sizes; (void)n_in; (void)out_size; (void)ws_size;

    const float* x      = (const float*)d_in[0];
    const float* hidden = (const float*)d_in[1];
    const float* rw1    = (const float*)d_in[14];
    const float* rb1    = (const float*)d_in[15];
    const float* rw3    = (const float*)d_in[16];
    const float* rb3    = (const float*)d_in[17];

    float* out   = (float*)d_out;
    float* means = (float*)d_ws;                 // 256 floats scratch

    mean_kernel<<<256, 256, 0, stream>>>(x, means);
    router_kernel<<<1, 64, 0, stream>>>(means, hidden, rw1, rb1, rw3, rb3,
                                        out + (size_t)4 * 64 * 65536);

    Params P;
    P.x = x;
    P.w1[0] = (const float*)d_in[2];  P.b1[0] = (const float*)d_in[3];
    P.w2[0] = (const float*)d_in[4];  P.b2[0] = (const float*)d_in[5];
    P.w1[1] = (const float*)d_in[6];  P.b1[1] = (const float*)d_in[7];
    P.w2[1] = (const float*)d_in[8];  P.b2[1] = (const float*)d_in[9];
    P.w1[2] = (const float*)d_in[10]; P.b1[2] = (const float*)d_in[11];
    P.w2[2] = (const float*)d_in[12]; P.b2[2] = (const float*)d_in[13];
    P.logit = out + (size_t)4 * 64 * 65536 + 64;
    P.out   = out;

    expert_kernel<<<512, 256, LDS_BYTES, stream>>>(P);
}
